// LocalRNN_80178449482213
// MI455X (gfx1250) — matrix operand order repaired and checked by reading, not compiled, not run on hardware
//
#include <hip/hip_runtime.h>
#include <hip/hip_bf16.h>

// ---------------------------------------------------------------------------
// LocalRNN fused WMMA implementation for gfx1250 (MI455X).
//   proj = pad(x) @ Wx      bf16 hi/lo split GEMM (fp32-class accuracy)
//   16-step scan            state resident in LDS as bf16 hi/lo (split kept:
//                           recurrent quantity, precision is LDS-cheap);
//                           Wy streamed from L2 as single bf16 (precision
//                           there costs 2x L2 bandwidth + 1.5x WMMAs).
// ---------------------------------------------------------------------------

typedef __attribute__((ext_vector_type(16))) __bf16 v16bf;
typedef __attribute__((ext_vector_type(8)))  float  v8f;

constexpr int kH  = 1024;           // hidden
constexpr int kS  = 2048;           // sequence
constexpr int kB  = 8;              // batch
constexpr int kW  = 16;             // window
constexpr int kSP = kS + kW - 1;    // padded sequence = 2063
constexpr int kM  = kB * kSP;       // proj rows = 16504

__device__ __forceinline__ v8f wmma_bf16(v16bf a, v16bf b, v8f c) {
    return __builtin_amdgcn_wmma_f32_16x16x32_bf16(
        /*neg_a=*/false, a, /*neg_b=*/false, b,
        /*c_mod=*/(short)0, c, /*reuse_a=*/false, /*reuse_b=*/false);
}

// A-fragment (16x32 bf16) element mapping: element (mr, kr) -> lane-group h, idx
__device__ __forceinline__ void afrag_map(int kr, int& h, int& idx) {
    h = kr >> 4; idx = kr & 15;
}

// ---------------------------------------------------------------------------
// Prep kernels: swizzle fp32 weights into WMMA B-fragment order
//   frag f = kc*(N/16)+nt ; within frag: [lane(32)][idx(16)] bf16
//   B-matrix 32x16 layout: lane = (n%16) + 16*(kr/16), idx = kr%16.
// Split (hi/lo) version for Wx, single-bf16 version for Wy.
// ---------------------------------------------------------------------------
__global__ void prep_weights_split(const float* __restrict__ Wsrc,
                                   __bf16* __restrict__ Whi,
                                   __bf16* __restrict__ Wlo,
                                   int K, int N) {
    size_t id = (size_t)blockIdx.x * blockDim.x + threadIdx.x;
    if (id >= (size_t)K * N) return;
    int k = (int)(id / N), n = (int)(id % N);
    float w = Wsrc[id];
    __bf16 hi = (__bf16)w;
    __bf16 lo = (__bf16)(w - (float)hi);
    int kc = k >> 5, kr = k & 31, nt = n >> 4, nr = n & 15;
    int lane = nr + ((kr >> 4) << 4);
    int idx  = kr & 15;
    size_t a = (((size_t)kc * (N >> 4) + nt) * 32 + lane) * 16 + idx;
    Whi[a] = hi;
    Wlo[a] = lo;
}

__global__ void prep_weights_single(const float* __restrict__ Wsrc,
                                    __bf16* __restrict__ Wb,
                                    int K, int N) {
    size_t id = (size_t)blockIdx.x * blockDim.x + threadIdx.x;
    if (id >= (size_t)K * N) return;
    int k = (int)(id / N), n = (int)(id % N);
    int kc = k >> 5, kr = k & 31, nt = n >> 4, nr = n & 15;
    int lane = nr + ((kr >> 4) << 4);
    int idx  = kr & 15;
    size_t a = (((size_t)kc * (N >> 4) + nt) * 32 + lane) * 16 + idx;
    Wb[a] = (__bf16)Wsrc[id];
}

// ---------------------------------------------------------------------------
// Kernel 1: proj[r][n] = xp[r] @ Wx (hi/lo split, fp32 accum).
// Tile: 16 rows x 1024 cols per workgroup (8 waves x 8 ntiles).
// ---------------------------------------------------------------------------
__global__ void proj_gemm(const float* __restrict__ x,
                          const __bf16* __restrict__ Wxh,
                          const __bf16* __restrict__ Wxl,
                          float* __restrict__ proj) {
    __shared__ __bf16 stage[2][2][512];   // [buf][hi/lo][lane*16+idx]
    const int tid  = threadIdx.x;
    const int lane = tid & 31;
    const int wv   = tid >> 5;
    const int r0   = blockIdx.x * 16;

    auto build = [&](int kc, int buf) {
        int mr = tid >> 5;       // 0..7
        int kr = tid & 31;       // coalesced along K within a row
        #pragma unroll
        for (int rep = 0; rep < 2; ++rep) {
            int m = mr + rep * 8;
            int r = r0 + m;
            float v = 0.0f;
            if (r < kM) {
                int t = r % kSP;
                int b = r / kSP;
                if (t >= kW - 1)
                    v = x[((size_t)b * kS + (t - (kW - 1))) * kH + kc * 32 + kr];
            }
            __bf16 hi = (__bf16)v;
            __bf16 lo = (__bf16)(v - (float)hi);
            int h, idx; afrag_map(kr, h, idx);
            int ln = m + 16 * h;
            stage[buf][0][ln * 16 + idx] = hi;
            stage[buf][1][ln * 16 + idx] = lo;
        }
    };

    build(0, 0);
    __syncthreads();

    v8f acc[8];
    v8f zero{};
    #pragma unroll
    for (int t = 0; t < 8; ++t) acc[t] = zero;

    for (int kc = 0; kc < 32; ++kc) {
        int buf = kc & 1;
        v16bf ah = *(const v16bf*)&stage[buf][0][lane * 16];
        v16bf al = *(const v16bf*)&stage[buf][1][lane * 16];
        #pragma unroll
        for (int t = 0; t < 8; ++t) {
            int nt = wv * 8 + t;
            size_t f = (((size_t)kc * 64 + nt) * 32 + lane) * 16;  // NT=64 for Wx
            v16bf bh = *(const v16bf*)(Wxh + f);
            v16bf bl = *(const v16bf*)(Wxl + f);
            acc[t] = wmma_bf16(ah, bh, acc[t]);
            acc[t] = wmma_bf16(ah, bl, acc[t]);
            acc[t] = wmma_bf16(al, bh, acc[t]);
        }
        if (kc + 1 < 32) build(kc + 1, (kc + 1) & 1);
        __syncthreads();
    }

    #pragma unroll
    for (int t = 0; t < 8; ++t) {
        int n0 = (wv * 8 + t) * 16;
        #pragma unroll
        for (int v = 0; v < 8; ++v) {
            int m = v + ((lane >> 4) << 3);
            int r = r0 + m;
            if (r < kM) proj[(size_t)r * kH + n0 + (lane & 15)] = acc[t][v];
        }
    }
}

// ---------------------------------------------------------------------------
// Kernel 2: 16-step recurrent scan. Each WG owns 32 sequence positions.
// State in LDS as bf16 hi/lo, double-buffered, in WMMA A-fragment order:
//   sm[buf*65536 + arr*32768 + ((mt*32+kc)*32+lane)*16+idx]   (256 KB LDS)
// Per wave: 8 cand/gate column-pairs processed in groups of 2 (A-frag reuse).
// One barrier per step; last step writes fp32 straight to d_out.
// ---------------------------------------------------------------------------
__global__ void scan_kernel(const float* __restrict__ proj,
                            const __bf16* __restrict__ Wyb,
                            const float* __restrict__ by,
                            float* __restrict__ out) {
    extern __shared__ __bf16 sm[];
    const int tid  = threadIdx.x;
    const int lane = tid & 31;
    const int wv   = tid >> 5;
    const int row0 = blockIdx.x * 32;
    const int b    = row0 / kS;
    const int srow = row0 % kS;

    // zero initial state (buffer 0): 128 KB
    {
        uint4* p = (uint4*)sm;
        uint4  z = {0u, 0u, 0u, 0u};
        for (int j = tid; j < 8192; j += 256) p[j] = z;
    }
    __syncthreads();

    int cur = 0;
    for (int i = 0; i < kW; ++i) {
        const bool last = (i == kW - 1);
        const __bf16* curh = sm + (size_t)cur * 65536;
        const __bf16* curl = curh + 32768;
        __bf16* nxth = sm + (size_t)(cur ^ 1) * 65536;
        __bf16* nxtl = nxth + 32768;

        for (int g = 0; g < 4; ++g) {
            const int Pbase = wv * 8 + g * 2;
            v8f acc[2][2][2];                 // [pair][cand/gate][mtile]
            v8f zero{};
            #pragma unroll
            for (int q = 0; q < 2; ++q)
                #pragma unroll
                for (int c = 0; c < 2; ++c)
                    #pragma unroll
                    for (int mt = 0; mt < 2; ++mt) acc[q][c][mt] = zero;

            for (int kc = 0; kc < 32; ++kc) {
                v16bf a0h = *(const v16bf*)(curh + (((size_t)kc)      * 32 + lane) * 16);
                v16bf a0l = *(const v16bf*)(curl + (((size_t)kc)      * 32 + lane) * 16);
                v16bf a1h = *(const v16bf*)(curh + (((size_t)32 + kc) * 32 + lane) * 16);
                v16bf a1l = *(const v16bf*)(curl + (((size_t)32 + kc) * 32 + lane) * 16);
                #pragma unroll
                for (int q = 0; q < 2; ++q) {
                    const int P = Pbase + q;
                    size_t fc = (((size_t)kc * 128 + P)      * 32 + lane) * 16;
                    size_t fg = (((size_t)kc * 128 + 64 + P) * 32 + lane) * 16;
                    v16bf bc = *(const v16bf*)(Wyb + fc);
                    v16bf bg = *(const v16bf*)(Wyb + fg);
                    // prefetch the Wy stream 4 K-chunks ahead (L2 latency;
                    // only 2 waves/SIMD resident) -> global_prefetch_b8
                    if (kc + 4 < 32) {
                        __builtin_prefetch(Wyb + fc + (size_t)4 * 65536, 0, 1);
                        __builtin_prefetch(Wyb + fg + (size_t)4 * 65536, 0, 1);
                    }
                    acc[q][0][0] = wmma_bf16(a0h, bc, acc[q][0][0]);
                    acc[q][0][0] = wmma_bf16(a0l, bc, acc[q][0][0]);
                    acc[q][0][1] = wmma_bf16(a1h, bc, acc[q][0][1]);
                    acc[q][0][1] = wmma_bf16(a1l, bc, acc[q][0][1]);
                    acc[q][1][0] = wmma_bf16(a0h, bg, acc[q][1][0]);
                    acc[q][1][0] = wmma_bf16(a0l, bg, acc[q][1][0]);
                    acc[q][1][1] = wmma_bf16(a1h, bg, acc[q][1][1]);
                    acc[q][1][1] = wmma_bf16(a1l, bg, acc[q][1][1]);
                }
            }

            // ---- epilogue: gate + state update, write next state ----
            #pragma unroll
            for (int q = 0; q < 2; ++q) {
                const int P  = Pbase + q;
                const int n0 = P * 16;
                const float bc  = by[n0 + (lane & 15)];
                const float bg  = by[kH + n0 + (lane & 15)];
                const int   kcE = n0 >> 5;                    // constant per pair
                const int   kr  = (n0 & 31) + (lane & 15);    // 0..31
                int h, idx; afrag_map(kr, h, idx);

                #pragma unroll
                for (int mt = 0; mt < 2; ++mt) {
                    #pragma unroll
                    for (int v = 0; v < 8; ++v) {
                        int mr   = v + ((lane >> 4) << 3);    // 0..15
                        int mloc = mt * 16 + mr;
                        float cand = acc[q][0][mt][v] + bc;
                        float glog = acc[q][1][mt][v] + bg;
                        float sg   = 1.0f / (1.0f + __expf(-glog));
                        float gate = fminf(fmaxf(1.2f * sg - 0.1f, 0.0f), 1.0f);
                        float inp  = proj[((size_t)b * kSP + srow + mloc + i) * kH
                                          + n0 + (lane & 15)];
                        size_t ea  = (((size_t)mt * 32 + kcE) * 32 + (mr + 16 * h)) * 16 + idx;
                        float oldv = (float)curh[ea] + (float)curl[ea];
                        float nv   = fmaxf(gate * (inp + cand) + (1.0f - gate) * oldv, 0.0f);
                        if (last) {
                            out[((size_t)b * kS + srow + mloc) * kH + n0 + (lane & 15)] = nv;
                        } else {
                            __bf16 hi = (__bf16)nv;
                            nxth[ea] = hi;
                            nxtl[ea] = (__bf16)(nv - (float)hi);
                        }
                    }
                }
            }
        }
        __syncthreads();
        cur ^= 1;
    }
}

// ---------------------------------------------------------------------------
// Launch: prep weights -> proj GEMM -> fused scan. All on `stream`.
// Workspace: proj f32 (67.6 MB) | Wx hi/lo bf16 (4 MB) | Wy bf16 (4 MB)
// ---------------------------------------------------------------------------
extern "C" void kernel_launch(void* const* d_in, const int* in_sizes, int n_in,
                              void* d_out, int out_size, void* d_ws, size_t ws_size,
                              hipStream_t stream) {
    (void)in_sizes; (void)n_in; (void)out_size; (void)ws_size;
    const float* x  = (const float*)d_in[0];
    const float* Wx = (const float*)d_in[1];
    const float* Wy = (const float*)d_in[2];
    const float* by = (const float*)d_in[3];
    float* out = (float*)d_out;

    char*  ws   = (char*)d_ws;
    float* proj = (float*)ws;
    size_t off  = (size_t)kM * kH * sizeof(float);          // 67,600,384 B
    __bf16* Wxh = (__bf16*)(ws + off); off += (size_t)kH * kH * 2;
    __bf16* Wxl = (__bf16*)(ws + off); off += (size_t)kH * kH * 2;
    __bf16* Wyb = (__bf16*)(ws + off);

    // 1) swizzle weights into WMMA B-fragment layout
    prep_weights_split<<<(kH * kH + 255) / 256, 256, 0, stream>>>(Wx, Wxh, Wxl, kH, kH);
    prep_weights_single<<<(kH * 2 * kH + 255) / 256, 256, 0, stream>>>(Wy, Wyb, kH, 2 * kH);

    // 2) proj = pad(x) @ Wx   (proj stays hot in 192 MB L2 for the scan)
    proj_gemm<<<(kM + 15) / 16, 256, 0, stream>>>(x, Wxh, Wxl, proj);

    // 3) fused 16-step recurrence, state resident in LDS (256 KB dynamic)
    scan_kernel<<<(kB * kS) / 32, 256, 262144, stream>>>(proj, Wyb, by, out);
}
